// _NonLocalBlockND_57913339019453
// MI455X (gfx1250) — matrix operand order repaired and checked by reading, not compiled, not run on hardware
//
#include <hip/hip_runtime.h>

// ---------------- problem constants ----------------
constexpr int BATCH = 4;
constexpr int CIN   = 256;   // C
constexpr int HT    = 96;
constexpr int WT    = 96;
constexpr int NPIX  = HT * WT;        // 9216
constexpr int CI    = 128;            // inner channels
constexpr int HP    = HT / 2;         // 48
constexpr int WP    = WT / 2;         // 48
constexpr int MP    = HP * WP;        // 2304 pooled positions

typedef __attribute__((ext_vector_type(16))) __bf16 bf16x16;
typedef __attribute__((ext_vector_type(8)))  float  f32x8;

// ---------------- WMMA helper ----------------
__device__ __forceinline__ f32x8 wmma_bf16(bf16x16 a, bf16x16 b, f32x8 c) {
    // D = A(16x32 bf16) * B(32x16 bf16) + C(16x16 f32)
    return __builtin_amdgcn_wmma_f32_16x16x32_bf16(
        /*neg_a=*/false, a, /*neg_b=*/false, b,
        /*c_mod=*/(short)0, c, /*reuse_a=*/false, /*reuse_b=*/false);
}

// A-fragment (16x32 bf16), source row-major with leading dim ld.
// Lane L: row = L&15, half = L>>4. VGPR v holds K = 16*half+2*v .. +1
__device__ __forceinline__ bf16x16 load_a_frag(const __bf16* p, int ld) {
    int lane = threadIdx.x & 31;
    int row  = lane & 15;
    int half = lane >> 4;
    const __bf16* rp = p + (size_t)row * ld;
    bf16x16 a;
#pragma unroll
    for (int v = 0; v < 8; ++v) {
        int kb = (half << 4) + (v << 1);
        a[2 * v]     = rp[kb];
        a[2 * v + 1] = rp[kb + 1];
    }
    return a;
}

// B-fragment (32x16 bf16) where source is "transposed": B(k,n) = p[n*ld + k]
// Lane L: col n = L&15, half = L>>4. VGPR v holds K = 16*half + 2v .. +1
__device__ __forceinline__ bf16x16 load_bT_frag(const __bf16* p, int ld) {
    int lane = threadIdx.x & 31;
    int n    = lane & 15;
    int half = lane >> 4;
    const __bf16* cp = p + (size_t)n * ld + (half << 4);
    bf16x16 b;
#pragma unroll
    for (int v = 0; v < 8; ++v) {
        b[2 * v]     = cp[2 * v];
        b[2 * v + 1] = cp[2 * v + 1];
    }
    return b;
}

// ---------------- conversion kernels ----------------
// x [B, C, N] f32 -> xT [B, N, C] bf16, LDS-tiled transpose (coalesced both sides)
__global__ __launch_bounds__(256) void transpose_cvt_kernel(const float* __restrict__ x,
                                                            __bf16* __restrict__ xT) {
    __shared__ float tile[32][33];
    int b  = blockIdx.z;
    int n0 = blockIdx.x * 32;
    int c0 = blockIdx.y * 32;
    int tn = threadIdx.x;        // 0..31
    int tc = threadIdx.y;        // 0..7
    const float* xb = x + (size_t)b * CIN * NPIX;
#pragma unroll
    for (int i = 0; i < 32; i += 8)
        tile[tc + i][tn] = xb[(size_t)(c0 + tc + i) * NPIX + n0 + tn];
    __syncthreads();
    __bf16* xtb = xT + (size_t)b * NPIX * CIN;
#pragma unroll
    for (int i = 0; i < 32; i += 8)
        xtb[(size_t)(n0 + tc + i) * CIN + c0 + tn] = (__bf16)tile[tn][tc + i];
}

__global__ void cvt_f32_to_bf16_kernel(const float* __restrict__ s, __bf16* __restrict__ d, int n) {
    int i = blockIdx.x * blockDim.x + threadIdx.x;
    if (i < n) d[i] = (__bf16)s[i];
}

// ---------------- projection GEMM: out[b,n,ci] = sum_c xT[b,n,c]*w[ci,c] + bias[ci] ----------------
// Weights staged once per block into LDS, pre-swizzled into B-fragment order:
//   wlds[((s*8 + t)*32 + lane) * 32B]  ->  lane-contiguous, conflict-free ds_load_b128 pairs.
__global__ __launch_bounds__(128) void conv_proj_kernel(
    const __bf16* __restrict__ xT,    // [B, N, C]
    const __bf16* __restrict__ wbf,   // [CI, C] row-major
    const float*  __restrict__ bias,  // [CI]
    __bf16* __restrict__ out)         // [B, N, CI] bf16
{
    __shared__ uint4 wlds[4096];      // 64 KB, shared by the 4 waves of this block

    // ---- cooperative stage: fragment (s,t), lane L=(n,h): w[(t*16+n)*CIN + s*32 + h*16 ..+15]
    {
        const uint4* wg = (const uint4*)wbf;   // weight row stride = CIN*2/16 = 32 uint4
        for (int u = threadIdx.x; u < 4096; u += 128) {
            int c = u & 1;
            int L = (u >> 1) & 31;
            int t = (u >> 6) & 7;
            int s = (u >> 9) & 7;
            int n = L & 15, h = L >> 4;
            wlds[u] = wg[(size_t)(t * 16 + n) * 32 + s * 4 + h * 2 + c];
        }
    }
    __syncthreads();

    int wave = blockIdx.x * (blockDim.x >> 5) + (threadIdx.x >> 5);
    int b  = wave / (NPIX / 16);
    int rt = wave % (NPIX / 16);
    int n0 = rt * 16;
    int lane = threadIdx.x & 31;
    int coln = lane & 15;
    int half = lane >> 4;

    const __bf16* aBase = xT + ((size_t)b * NPIX + n0) * CIN;
    const __bf16* wfr   = (const __bf16*)wlds;

    f32x8 acc[8];
#pragma unroll
    for (int t = 0; t < 8; ++t) {
        float bv = bias[t * 16 + coln];   // C init = bias broadcast along rows
#pragma unroll
        for (int r = 0; r < 8; ++r) acc[t][r] = bv;
    }

    constexpr int KSTEPS = CIN / 32;   // 8
    bf16x16 a = load_a_frag(aBase, CIN);
#pragma unroll
    for (int s = 0; s < KSTEPS; ++s) {
        bf16x16 an;
        if (s + 1 < KSTEPS)            // prefetch next A tile while matrix pipe is busy
            an = load_a_frag(aBase + (s + 1) * 32, CIN);
#pragma unroll
        for (int t = 0; t < 8; ++t) {
            bf16x16 bb = *(const bf16x16*)(wfr + (((s * 8 + t) * 32 + lane) << 4));
            acc[t] = wmma_bf16(a, bb, acc[t]);
        }
        if (s + 1 < KSTEPS) a = an;
    }

    __bf16* oBase = out + ((size_t)b * NPIX + n0) * CI;
#pragma unroll
    for (int t = 0; t < 8; ++t)
#pragma unroll
        for (int r = 0; r < 8; ++r)
            oBase[(size_t)(r + 8 * half) * CI + t * 16 + coln] = (__bf16)acc[t][r];
}

// ---------------- 2x2 maxpool: [B,N,CI] -> pooled with caller-chosen layout ----------------
// out[b ... m*sm + ci*sc]; phi uses (sm=CI, sc=1) -> [B,M,CI]; g uses (sm=1, sc=MP) -> [B,CI,M]
__global__ void pool_kernel(const __bf16* __restrict__ conv, __bf16* __restrict__ pooled,
                            int sm, int sc) {
    size_t i = (size_t)blockIdx.x * blockDim.x + threadIdx.x;
    size_t total = (size_t)BATCH * MP * CI;
    if (i >= total) return;
    int ci = (int)(i % CI);
    size_t rem = i / CI;
    int m = (int)(rem % MP);
    int b = (int)(rem / MP);
    int pr = m / WP, pc = m % WP;
    int n00 = (2 * pr) * WT + 2 * pc;
    const __bf16* base = conv + (size_t)b * NPIX * CI;
    float v0 = (float)base[(size_t)(n00)          * CI + ci];
    float v1 = (float)base[(size_t)(n00 + 1)      * CI + ci];
    float v2 = (float)base[(size_t)(n00 + WT)     * CI + ci];
    float v3 = (float)base[(size_t)(n00 + WT + 1) * CI + ci];
    float v = fmaxf(fmaxf(v0, v1), fmaxf(v2, v3));
    pooled[(size_t)b * MP * CI + (size_t)m * sm + (size_t)ci * sc] = (__bf16)v;
}

// ---------------- fused flash attention ----------------
// y[b,n,:] = softmax_m( theta[b,n,:] . phiP[b,m,:] ) @ g
// theta: [B,N,CI] (Q rows), phiP: [B,M,CI] (K rows), gT: [B,CI,M] (V channel-major)
__global__ __launch_bounds__(128) void attention_kernel(
    const __bf16* __restrict__ theta,
    const __bf16* __restrict__ phiP,
    const __bf16* __restrict__ gT,
    __bf16* __restrict__ yout)         // [B, N, CI]
{
    __shared__ __bf16 pbuf[4][16 * 34];   // per-wave 16x32 P staging, ld=34 (bank-skewed)

    int wid  = threadIdx.x >> 5;
    int wave = blockIdx.x * 4 + wid;
    int b  = wave / (NPIX / 16);
    int rt = wave % (NPIX / 16);
    int n0 = rt * 16;
    int lane = threadIdx.x & 31;
    int coln = lane & 15;
    int half = lane >> 4;

    // preload Q tile (16 x 128) as 4 A-fragments
    const __bf16* qBase = theta + ((size_t)b * NPIX + n0) * CI;
    bf16x16 qf[4];
#pragma unroll
    for (int kq = 0; kq < 4; ++kq) qf[kq] = load_a_frag(qBase + kq * 32, CI);

    float rmax[8], rsum[8];
#pragma unroll
    for (int r = 0; r < 8; ++r) { rmax[r] = -1e30f; rsum[r] = 0.0f; }
    f32x8 yacc[8];
#pragma unroll
    for (int t = 0; t < 8; ++t)
#pragma unroll
        for (int r = 0; r < 8; ++r) yacc[t][r] = 0.0f;

    __bf16* P = &pbuf[wid][0];
    const __bf16* phiB = phiP + (size_t)b * MP * CI;
    const __bf16* gTb  = gT   + (size_t)b * CI * MP;

    for (int m0 = 0; m0 < MP; m0 += 32) {
        // prefetch next chunk of K and V into cache (global_prefetch_b8)
        if (m0 + 32 < MP) {
            __builtin_prefetch(phiB + (size_t)(m0 + 32) * CI + lane * 32, 0, 1);
            __builtin_prefetch(gTb + (size_t)(lane * 4) * MP + m0 + 32, 0, 1);
        }

        // ---- S = Q * K^T for 32 keys: batch all 8 K-fragments, then 8 WMMAs ----
        bf16x16 kb[8];
#pragma unroll
        for (int kq = 0; kq < 4; ++kq) {
            kb[2 * kq]     = load_bT_frag(phiB + (size_t)m0        * CI + kq * 32, CI);
            kb[2 * kq + 1] = load_bT_frag(phiB + (size_t)(m0 + 16) * CI + kq * 32, CI);
        }
        f32x8 s0, s1;
#pragma unroll
        for (int r = 0; r < 8; ++r) { s0[r] = 0.0f; s1[r] = 0.0f; }
#pragma unroll
        for (int kq = 0; kq < 4; ++kq) {
            s0 = wmma_bf16(qf[kq], kb[2 * kq],     s0);
            s1 = wmma_bf16(qf[kq], kb[2 * kq + 1], s1);
        }

        // ---- issue V fragment loads now; softmax VALU work hides their latency ----
        // V channel-major: B(k=m', n=ci') = gT[(ci0+n)*MP + m0+k] -> contiguous per lane
        bf16x16 vf[8];
#pragma unroll
        for (int t = 0; t < 8; ++t)
            vf[t] = load_bT_frag(gTb + (size_t)(t * 16) * MP + m0, MP);

        // ---- chunk row max (reduce across 16-lane group) ----
        float cm[8];
#pragma unroll
        for (int r = 0; r < 8; ++r) cm[r] = fmaxf(s0[r], s1[r]);
#pragma unroll
        for (int d = 1; d < 16; d <<= 1)
#pragma unroll
            for (int r = 0; r < 8; ++r) cm[r] = fmaxf(cm[r], __shfl_xor(cm[r], d, 32));

        // ---- online rescale of running stats + accumulators ----
        float scale[8];
#pragma unroll
        for (int r = 0; r < 8; ++r) {
            float nm = fmaxf(rmax[r], cm[r]);
            scale[r] = __expf(rmax[r] - nm);
            rmax[r]  = nm;
            rsum[r] *= scale[r];
        }
#pragma unroll
        for (int t = 0; t < 8; ++t)
#pragma unroll
            for (int r = 0; r < 8; ++r) yacc[t][r] *= scale[r];

        // ---- P = exp(S - rmax): D-layout -> LDS, track partial row sums ----
        float ps[8];
#pragma unroll
        for (int r = 0; r < 8; ++r) {
            float p0 = __expf(s0[r] - rmax[r]);
            float p1 = __expf(s1[r] - rmax[r]);
            int row = r + 8 * half;
            P[row * 34 + coln]      = (__bf16)p0;
            P[row * 34 + 16 + coln] = (__bf16)p1;
            ps[r] = p0 + p1;
        }
#pragma unroll
        for (int d = 1; d < 16; d <<= 1)
#pragma unroll
            for (int r = 0; r < 8; ++r) ps[r] += __shfl_xor(ps[r], d, 32);
#pragma unroll
        for (int r = 0; r < 8; ++r) rsum[r] += ps[r];

        // LDS stores must land before A-fragment reads (same wave, DScnt fence)
        asm volatile("s_wait_dscnt 0" ::: "memory");

        // ---- y += P * V (V fragments already in flight/registers) ----
        bf16x16 pf = load_a_frag(P, 34);
#pragma unroll
        for (int t = 0; t < 8; ++t)
            yacc[t] = wmma_bf16(pf, vf[t], yacc[t]);
        asm volatile("s_wait_dscnt 0" ::: "memory");
    }

    // ---- normalize and store y (bf16, [B,N,CI]) ----
    float inv[8];
#pragma unroll
    for (int r = 0; r < 8; ++r) inv[r] = 1.0f / rsum[r];
    __bf16* yB = yout + ((size_t)b * NPIX + n0) * CI;
#pragma unroll
    for (int t = 0; t < 8; ++t)
#pragma unroll
        for (int r = 0; r < 8; ++r)
            yB[(size_t)(r + 8 * half) * CI + t * 16 + coln] = (__bf16)(yacc[t][r] * inv[r]);
}

// ---------------- final conv + bias + residual + relu ----------------
// z[b,c,n] = relu( sum_ci y[b,n,ci]*w_w[c,ci] + w_b[c] + x[b,c,n] )
// Full 256x128 weight staged per block in LDS, pre-swizzled B-fragment order.
__global__ __launch_bounds__(128) void final_conv_kernel(
    const __bf16* __restrict__ ybf,   // [B, N, CI]
    const __bf16* __restrict__ wwbf,  // [C, CI] row-major
    const float*  __restrict__ wb,    // [C]
    const float*  __restrict__ x,     // [B, C, N]
    float* __restrict__ out)          // [B, C, N]
{
    __shared__ uint4 wlds[4096];      // 64 KB: fragments [s(4)][tt(16)][lane(32)][32B]
    {
        const uint4* wg = (const uint4*)wwbf;  // row stride = CI*2/16 = 16 uint4
        for (int u = threadIdx.x; u < 4096; u += 128) {
            int c  = u & 1;
            int L  = (u >> 1) & 31;
            int tt = (u >> 6) & 15;
            int s  = (u >> 10) & 3;
            int n = L & 15, h = L >> 4;
            wlds[u] = wg[(size_t)(tt * 16 + n) * 16 + s * 4 + h * 2 + c];
        }
    }
    __syncthreads();

    int wid  = threadIdx.x >> 5;
    int wave = blockIdx.x * 4 + wid;
    int ct  = wave & 3;                 // 4 column tiles of 64 -> C=256 (wid == ct here)
    int tmp = wave >> 2;
    int rt  = tmp % (NPIX / 16);
    int b   = tmp / (NPIX / 16);
    int n0 = rt * 16;
    int c0 = ct * 64;
    int lane = threadIdx.x & 31;
    int coln = lane & 15;
    int half = lane >> 4;

    f32x8 acc[4];
#pragma unroll
    for (int t = 0; t < 4; ++t) {
        float bv = wb[c0 + t * 16 + coln];
#pragma unroll
        for (int r = 0; r < 8; ++r) acc[t][r] = bv;
    }

    const __bf16* aBase = ybf + ((size_t)b * NPIX + n0) * CI;
    const __bf16* wfr   = (const __bf16*)wlds;

    constexpr int KSTEPS = CI / 32;    // 4
    bf16x16 a = load_a_frag(aBase, CI);
#pragma unroll
    for (int s = 0; s < KSTEPS; ++s) {
        bf16x16 an;
        if (s + 1 < KSTEPS)
            an = load_a_frag(aBase + (s + 1) * 32, CI);
#pragma unroll
        for (int t = 0; t < 4; ++t) {
            int tt = ct * 4 + t;
            bf16x16 bb = *(const bf16x16*)(wfr + (((s * 16 + tt) * 32 + lane) << 4));
            acc[t] = wmma_bf16(a, bb, acc[t]);
        }
        if (s + 1 < KSTEPS) a = an;
    }

    const float* xB = x   + (size_t)b * CIN * NPIX;
    float*       oB = out + (size_t)b * CIN * NPIX;
#pragma unroll
    for (int t = 0; t < 4; ++t) {
        int c = c0 + t * 16 + coln;
#pragma unroll
        for (int r = 0; r < 8; ++r) {
            int n = n0 + r + 8 * half;
            float v = acc[t][r] + xB[(size_t)c * NPIX + n];
            oB[(size_t)c * NPIX + n] = v > 0.0f ? v : 0.0f;
        }
    }
}

// ---------------- host side ----------------
extern "C" void kernel_launch(void* const* d_in, const int* in_sizes, int n_in,
                              void* d_out, int out_size, void* d_ws, size_t ws_size,
                              hipStream_t stream) {
    const float* x       = (const float*)d_in[0];
    const float* g_w     = (const float*)d_in[1];
    const float* g_b     = (const float*)d_in[2];
    const float* theta_w = (const float*)d_in[3];
    const float* theta_b = (const float*)d_in[4];
    const float* phi_w   = (const float*)d_in[5];
    const float* phi_b   = (const float*)d_in[6];
    const float* w_w     = (const float*)d_in[7];
    const float* w_b     = (const float*)d_in[8];
    float* out = (float*)d_out;

    char* ws = (char*)d_ws;
    size_t off = 0;
    auto carve = [&](size_t bytes) -> char* {
        char* p = ws + off;
        off = (off + bytes + 255) & ~(size_t)255;
        return p;
    };

    __bf16* xT_bf   = (__bf16*)carve((size_t)BATCH * NPIX * CIN * 2);
    __bf16* gw_bf   = (__bf16*)carve((size_t)CI * CIN * 2);
    __bf16* thw_bf  = (__bf16*)carve((size_t)CI * CIN * 2);
    __bf16* phw_bf  = (__bf16*)carve((size_t)CI * CIN * 2);
    __bf16* ww_bf   = (__bf16*)carve((size_t)CIN * CI * 2);
    __bf16* theta_bf= (__bf16*)carve((size_t)BATCH * NPIX * CI * 2);
    __bf16* phic_bf = (__bf16*)carve((size_t)BATCH * NPIX * CI * 2);
    __bf16* gc_bf   = (__bf16*)carve((size_t)BATCH * NPIX * CI * 2);
    __bf16* phip_bf = (__bf16*)carve((size_t)BATCH * MP * CI * 2);   // [B,M,CI]
    __bf16* gpT_bf  = (__bf16*)carve((size_t)BATCH * CI * MP * 2);   // [B,CI,M]
    __bf16* y_bf    = (__bf16*)carve((size_t)BATCH * NPIX * CI * 2);
    (void)ws_size; (void)n_in; (void)in_sizes; (void)out_size;

    // 1) tiled transpose+convert x, convert weights to bf16
    {
        dim3 tg(NPIX / 32, CIN / 32, BATCH);
        transpose_cvt_kernel<<<tg, dim3(32, 8, 1), 0, stream>>>(x, xT_bf);
        int nw = CI * CIN;
        cvt_f32_to_bf16_kernel<<<(nw + 255) / 256, 256, 0, stream>>>(g_w,     gw_bf,  nw);
        cvt_f32_to_bf16_kernel<<<(nw + 255) / 256, 256, 0, stream>>>(theta_w, thw_bf, nw);
        cvt_f32_to_bf16_kernel<<<(nw + 255) / 256, 256, 0, stream>>>(phi_w,   phw_bf, nw);
        cvt_f32_to_bf16_kernel<<<(nw + 255) / 256, 256, 0, stream>>>(w_w,     ww_bf,  nw);
    }

    // 2) projection GEMMs (theta, phi, g) — WMMA bf16 with LDS-resident weights
    {
        int waves  = BATCH * (NPIX / 16);
        int blocks = waves / 4;
        conv_proj_kernel<<<blocks, 128, 0, stream>>>(xT_bf, thw_bf, theta_b, theta_bf);
        conv_proj_kernel<<<blocks, 128, 0, stream>>>(xT_bf, phw_bf, phi_b,   phic_bf);
        conv_proj_kernel<<<blocks, 128, 0, stream>>>(xT_bf, gw_bf,  g_b,     gc_bf);
    }

    // 3) 2x2 max pool: phi -> [B,M,CI], g -> [B,CI,M] (channel-major for contiguous V frags)
    {
        size_t total = (size_t)BATCH * MP * CI;
        pool_kernel<<<(unsigned)((total + 255) / 256), 256, 0, stream>>>(phic_bf, phip_bf, CI, 1);
        pool_kernel<<<(unsigned)((total + 255) / 256), 256, 0, stream>>>(gc_bf,   gpT_bf,  1,  MP);
    }

    // 4) fused flash attention
    {
        int waves  = BATCH * (NPIX / 16);
        int blocks = waves / 4;
        attention_kernel<<<blocks, 128, 0, stream>>>(theta_bf, phip_bf, gpT_bf, y_bf);
    }

    // 5) final conv + bias + residual + relu with LDS-resident weights
    {
        int waves  = BATCH * (NPIX / 16) * (CIN / 64);
        int blocks = waves / 4;
        final_conv_kernel<<<blocks, 128, 0, stream>>>(y_bf, ww_bf, w_b, x, out);
    }
}
